// ASTGCNBlock_38482906972701
// MI455X (gfx1250) — compile-verified, matrix operand order checked by reading, not run on hardware
//
#include <hip/hip_runtime.h>
#include <hip/hip_bf16.h>

typedef __bf16 bf16_t;
typedef __attribute__((ext_vector_type(16))) __bf16 v16bf;
typedef __attribute__((ext_vector_type(8)))  __bf16 v8bf;
typedef __attribute__((ext_vector_type(8)))  float  v8f;

#define BDIM 8
#define CDIM 64
#define NDIM 1024
#define TDIM 12
#define MDIM (BDIM * TDIM)          // 96
#define NC   (NDIM * CDIM)          // 65536
#define NN   (NDIM * NDIM)          // 1048576

// ---------------------------------------------------------------------------
// WMMA fragment helpers (CDNA5 ISA 7.12.2 layouts, wave32)
// ---------------------------------------------------------------------------

// 16x32 (rows x K) fragment from a row-major matrix, ld in elements.
// Serves as the A operand, or as the B operand when the source holds B^T
// (row r of the source = column r of B). Two contiguous 16B loads per lane.
__device__ __forceinline__ v16bf load_frag_row(const bf16_t* __restrict__ p, int ld) {
    const int lane = threadIdx.x & 31;
    const int r    = lane & 15;
    const int half = lane >> 4;
    const bf16_t* q = p + (size_t)r * ld + half * 8;
    v8bf lo = *reinterpret_cast<const v8bf*>(q);
    v8bf hi = *reinterpret_cast<const v8bf*>(q + 16);
    v16bf f;
#pragma unroll
    for (int i = 0; i < 8; ++i) { f[i] = lo[i]; f[i + 8] = hi[i]; }
    return f;
}

__device__ __forceinline__ v8f wmma_bf16(v16bf a, v16bf b, v8f c) {
    return __builtin_amdgcn_wmma_f32_16x16x32_bf16(false, a, false, b, (short)0, c,
                                                   false, false);
}

// C/D layout: vgpr r, lane: col = lane&15, row = r + 8*(lane>>4)
__device__ __forceinline__ void store_acc_bf16(bf16_t* p, int ld, v8f acc) {
    const int lane  = threadIdx.x & 31;
    const int col   = lane & 15;
    const int rbase = (lane >> 4) * 8;
#pragma unroll
    for (int r = 0; r < 8; ++r)
        p[(size_t)(rbase + r) * ld + col] = (bf16_t)acc[r];
}

// ---------------------------------------------------------------------------
// dtype conversion helpers
// ---------------------------------------------------------------------------
__global__ void k_f32_to_bf16(const float* __restrict__ s, bf16_t* __restrict__ d, int n) {
    int i = blockIdx.x * 256 + threadIdx.x;
    if (i < n) d[i] = (bf16_t)s[i];
}

// 64x64 transpose + convert (for W1/W2 so B-columns become contiguous rows)
__global__ void k_f32_to_bf16_T64(const float* __restrict__ s, bf16_t* __restrict__ d) {
    int i = blockIdx.x * 256 + threadIdx.x;   // 4096 elements
    int r = i >> 6, c = i & 63;
    d[c * 64 + r] = (bf16_t)s[i];
}

// ---------------------------------------------------------------------------
// 1) temporal attention: lhs[b,t] = sum_{n,c} x[b,c,n,t]*U3[c]*U1[n]
// ---------------------------------------------------------------------------
__global__ void k_temporal_lhs(const float* __restrict__ x, const float* __restrict__ U3,
                               const float* __restrict__ U1, float* __restrict__ lhs) {
    const int bt = blockIdx.x;                // 96
    const int b = bt / TDIM, t = bt % TDIM;
    float acc = 0.f;
    for (int n = threadIdx.x; n < NDIM; n += 256) {
        const float* xp = x + ((size_t)b * CDIM * NDIM + n) * TDIM + t;
        float s = 0.f;
#pragma unroll 8
        for (int c = 0; c < CDIM; ++c) s += xp[(size_t)c * NDIM * TDIM] * U3[c];
        acc += s * U1[n];
    }
    __shared__ float red[256];
    red[threadIdx.x] = acc;
    __syncthreads();
    for (int s = 128; s > 0; s >>= 1) {
        if (threadIdx.x < s) red[threadIdx.x] += red[threadIdx.x + s];
        __syncthreads();
    }
    if (threadIdx.x == 0) lhs[bt] = red[0];
}

// 2) E[b,i,k] = softmax_k( sum_j Ve[i,j]*sigmoid(lhs[b,j]*lhs[b,k]+be[j,k]) )
__global__ void k_temporal_E(const float* __restrict__ lhs, const float* __restrict__ be,
                             const float* __restrict__ Ve, float* __restrict__ E) {
    const int b = blockIdx.x;                 // 8, blockDim = 160
    __shared__ float Esh[TDIM][TDIM];
    if (threadIdx.x < 144) {
        const int i = threadIdx.x / TDIM, k = threadIdx.x % TDIM;
        float a = 0.f;
        for (int j = 0; j < TDIM; ++j) {
            float p = lhs[b * TDIM + j] * lhs[b * TDIM + k] + be[j * TDIM + k];
            a += Ve[i * TDIM + j] * (1.f / (1.f + __expf(-p)));
        }
        Esh[i][k] = a;
    }
    __syncthreads();
    if (threadIdx.x < TDIM) {
        const int r = threadIdx.x;
        float mx = Esh[r][0];
        for (int k = 1; k < TDIM; ++k) mx = fmaxf(mx, Esh[r][k]);
        float ex[TDIM], sm = 0.f;
        for (int k = 0; k < TDIM; ++k) { ex[k] = __expf(Esh[r][k] - mx); sm += ex[k]; }
        const float inv = 1.f / sm;
        for (int k = 0; k < TDIM; ++k) E[(b * TDIM + r) * TDIM + k] = ex[k] * inv;
    }
}

// 3) x_att[b,t,n,c] = sum_u E[b,t,u] * x[b,c,n,u]
//    stores BOTH layouts: xatt[m][n][c] (A operand) and xattT[m][c][n] (B operand)
__global__ void k_temporal_apply(const float* __restrict__ x, const float* __restrict__ E,
                                 bf16_t* __restrict__ xatt, bf16_t* __restrict__ xattT) {
    const size_t idx = (size_t)blockIdx.x * 256 + threadIdx.x;  // over B*N*C
    const int c = idx & 63;
    const int n = (idx >> 6) & 1023;
    const int b = (int)(idx >> 16);
    __shared__ float Esh[TDIM * TDIM];
    if (threadIdx.x < 144) Esh[threadIdx.x] = E[b * 144 + threadIdx.x];
    __syncthreads();
    float xv[TDIM];
    const float* xp = x + (((size_t)b * CDIM + c) * NDIM + n) * TDIM;
#pragma unroll
    for (int u = 0; u < TDIM; ++u) xv[u] = xp[u];
#pragma unroll
    for (int t = 0; t < TDIM; ++t) {
        float a = 0.f;
#pragma unroll
        for (int u = 0; u < TDIM; ++u) a += Esh[t * TDIM + u] * xv[u];
        const size_t m = (size_t)(b * TDIM + t);
        const bf16_t v = (bf16_t)a;
        xatt [m * NC + (size_t)n * CDIM + c] = v;
        xattT[m * NC + (size_t)c * NDIM + n] = v;
    }
}

// ---------------------------------------------------------------------------
// 4) l = x_s@W1, r = x_s@W2   (M=96*1024, K=64, N=64) -- WMMA
// ---------------------------------------------------------------------------
__global__ void k_gemm_lr(const bf16_t* __restrict__ xatt, const bf16_t* __restrict__ W1t,
                          const bf16_t* __restrict__ W2t, bf16_t* __restrict__ l,
                          bf16_t* __restrict__ r) {
    const int wave = threadIdx.x >> 5;
    const size_t row0 = ((size_t)blockIdx.x * 8 + wave) * 16;
    v8f accL[4], accR[4];
#pragma unroll
    for (int j = 0; j < 4; ++j) { accL[j] = (v8f){}; accR[j] = (v8f){}; }
#pragma unroll
    for (int ks = 0; ks < 2; ++ks) {
        v16bf a = load_frag_row(xatt + row0 * CDIM + ks * 32, CDIM);
#pragma unroll
        for (int j = 0; j < 4; ++j) {
            v16bf b1 = load_frag_row(W1t + j * 16 * 64 + ks * 32, 64);
            v16bf b2 = load_frag_row(W2t + j * 16 * 64 + ks * 32, 64);
            accL[j] = wmma_bf16(a, b1, accL[j]);
            accR[j] = wmma_bf16(a, b2, accR[j]);
        }
    }
#pragma unroll
    for (int j = 0; j < 4; ++j) {
        store_acc_bf16(l + row0 * CDIM + j * 16, CDIM, accL[j]);
        store_acc_bf16(r + row0 * CDIM + j * 16, CDIM, accR[j]);
    }
}

// 5) sigT[m,k,n] = sigmoid( r[m,k,:]·l[m,n,:] + bs[n,k] )  -- WMMA
//    (prod^T computed directly so the sigmoid output lands k-major; columns of
//     the next GEMM's B operand become contiguous rows -> no LDS transpose)
__global__ void k_prod_sigT(const bf16_t* __restrict__ l, const bf16_t* __restrict__ r,
                            const float* __restrict__ bs, bf16_t* __restrict__ sigT) {
    const int m   = blockIdx.x;                              // 96
    const int k0  = (blockIdx.y * 8 + (threadIdx.x >> 5)) * 16;  // row tile of sigT
    const int n0g = blockIdx.z * 64;                         // col group of 64
    const bf16_t* lm = l + (size_t)m * NC;
    const bf16_t* rm = r + (size_t)m * NC;
    v8f acc[4];
#pragma unroll
    for (int j = 0; j < 4; ++j) acc[j] = (v8f){};
#pragma unroll
    for (int ks = 0; ks < 2; ++ks) {
        v16bf a = load_frag_row(rm + (size_t)k0 * CDIM + ks * 32, CDIM);
#pragma unroll
        for (int j = 0; j < 4; ++j) {
            v16bf b = load_frag_row(lm + (size_t)(n0g + j * 16) * CDIM + ks * 32, CDIM);
            acc[j] = wmma_bf16(a, b, acc[j]);
        }
    }
    const int lane = threadIdx.x & 31;
    const int col = lane & 15, rbase = (lane >> 4) * 8;
    bf16_t* outb = sigT + (size_t)m * NN + (size_t)k0 * NDIM + n0g;
#pragma unroll
    for (int j = 0; j < 4; ++j)
#pragma unroll
        for (int rr = 0; rr < 8; ++rr) {
            const int k = k0 + rbase + rr;             // sigT row
            const int n = n0g + j * 16 + col;          // sigT col
            const float v = acc[j][rr] + bs[(size_t)n * NDIM + k];
            outb[(size_t)(rbase + rr) * NDIM + j * 16 + col] =
                (bf16_t)(1.f / (1.f + __expf(-v)));
        }
}

// 6) S[m,i,k] = sum_j Vs[i,j]*sigT[m,k,j]  (206 GFLOP GEMM; all-b128 loads,
//    4 independent accumulators per wave -> no WMMA RAW chain)
__global__ void k_spatial_S(const bf16_t* __restrict__ Vs_bf, const bf16_t* __restrict__ sigT,
                            bf16_t* __restrict__ S) {
    const int m   = blockIdx.x;                              // 96
    const int i0  = (blockIdx.y * 8 + (threadIdx.x >> 5)) * 16;
    const int k0g = blockIdx.z * 64;
    const bf16_t* st = sigT + (size_t)m * NN;
    v8f acc[4];
#pragma unroll
    for (int j = 0; j < 4; ++j) acc[j] = (v8f){};
    for (int ks = 0; ks < 32; ++ks) {
        const int j0 = ks * 32;
        v16bf a = load_frag_row(Vs_bf + (size_t)i0 * NDIM + j0, NDIM);
        if (ks + 1 < 32)
            __builtin_prefetch(Vs_bf + (size_t)i0 * NDIM + j0 + 32, 0, 1);
#pragma unroll
        for (int j = 0; j < 4; ++j) {
            v16bf b = load_frag_row(st + (size_t)(k0g + j * 16) * NDIM + j0, NDIM);
            acc[j] = wmma_bf16(a, b, acc[j]);
        }
    }
#pragma unroll
    for (int j = 0; j < 4; ++j)
        store_acc_bf16(S + (size_t)m * NN + (size_t)i0 * NDIM + k0g + j * 16, NDIM, acc[j]);
}

// 7) softmax over last dim of S (rows of 1024, bf16 in place, f32 math)
__global__ void k_softmax_S(bf16_t* __restrict__ S) {
    bf16_t* p = S + (size_t)blockIdx.x * NDIM;
    const int t = threadIdx.x;                  // 128 threads
    float v[8];
#pragma unroll
    for (int j = 0; j < 8; ++j) v[j] = (float)p[t + j * 128];
    float mx = v[0];
#pragma unroll
    for (int j = 1; j < 8; ++j) mx = fmaxf(mx, v[j]);
    __shared__ float red[128];
    red[t] = mx; __syncthreads();
    for (int s = 64; s > 0; s >>= 1) {
        if (t < s) red[t] = fmaxf(red[t], red[t + s]);
        __syncthreads();
    }
    mx = red[0]; __syncthreads();
    float sm = 0.f;
#pragma unroll
    for (int j = 0; j < 8; ++j) { v[j] = __expf(v[j] - mx); sm += v[j]; }
    red[t] = sm; __syncthreads();
    for (int s = 64; s > 0; s >>= 1) {
        if (t < s) red[t] += red[t + s];
        __syncthreads();
    }
    const float inv = 1.f / red[0];
#pragma unroll
    for (int j = 0; j < 8; ++j) p[t + j * 128] = (bf16_t)(v[j] * inv);
}

// 8) x2[b,c,n,t] = (S[m]@x_s[m])[n,c]   -- WMMA, B from xattT (contiguous)
__global__ void k_spatial_apply(const bf16_t* __restrict__ S, const bf16_t* __restrict__ xattT,
                                float* __restrict__ x2) {
    const int m  = blockIdx.x;                 // 96
    const int i0 = (blockIdx.y * 8 + (threadIdx.x >> 5)) * 16;
    const bf16_t* Sm = S + (size_t)m * NN;
    const bf16_t* xT = xattT + (size_t)m * NC;
    v8f acc[4];
#pragma unroll
    for (int j = 0; j < 4; ++j) acc[j] = (v8f){};
    for (int ks = 0; ks < 32; ++ks) {
        v16bf a = load_frag_row(Sm + (size_t)i0 * NDIM + ks * 32, NDIM);
#pragma unroll
        for (int j = 0; j < 4; ++j) {
            v16bf b = load_frag_row(xT + (size_t)(j * 16) * NDIM + ks * 32, NDIM);
            acc[j] = wmma_bf16(a, b, acc[j]);
        }
    }
    const int b = m / TDIM, t = m % TDIM;
    const int lane = threadIdx.x & 31;
    const int col = lane & 15, rbase = (lane >> 4) * 8;
#pragma unroll
    for (int j = 0; j < 4; ++j)
#pragma unroll
        for (int rr = 0; rr < 8; ++rr) {
            const int n = i0 + rbase + rr, c = j * 16 + col;
            x2[(((size_t)b * CDIM + c) * NDIM + n) * TDIM + t] = acc[j][rr];
        }
}

// 9) temporal conv (1,3) pad (0,1) + relu; output c-major hT[m][c][n] (bf16)
__global__ void k_tconv(const float* __restrict__ x2, const float* __restrict__ w,
                        const float* __restrict__ tb, bf16_t* __restrict__ hT) {
    const int b = blockIdx.x >> 10;            // grid.x = B*N
    const int n = blockIdx.x & 1023;
    __shared__ float xt[CDIM][TDIM];
    for (int e = threadIdx.x; e < CDIM * TDIM; e += 256) {
        const int c = e / TDIM, t = e % TDIM;
        xt[c][t] = x2[(((size_t)b * CDIM + c) * NDIM + n) * TDIM + t];
    }
    __syncthreads();
    for (int e = threadIdx.x; e < CDIM * TDIM; e += 256) {
        const int co = e / TDIM, t = e % TDIM;
        float a = tb[co];
        const float* wp = w + co * CDIM * 3;
#pragma unroll 8
        for (int c = 0; c < CDIM; ++c) {
            const float xm1 = (t > 0) ? xt[c][t - 1] : 0.f;
            const float xp1 = (t < TDIM - 1) ? xt[c][t + 1] : 0.f;
            a += wp[c * 3 + 0] * xm1 + wp[c * 3 + 1] * xt[c][t] + wp[c * 3 + 2] * xp1;
        }
        hT[((size_t)(b * TDIM + t)) * NC + (size_t)co * NDIM + n] = (bf16_t)fmaxf(a, 0.f);
    }
}

// 10) graph conv: hg[m,i,c] = sum_j adj[i,j]*hT[m,c,j]   -- WMMA
__global__ void k_graph(const bf16_t* __restrict__ adj_bf, const bf16_t* __restrict__ hT,
                        bf16_t* __restrict__ hg) {
    const int m  = blockIdx.x;                 // 96
    const int i0 = (blockIdx.y * 8 + (threadIdx.x >> 5)) * 16;
    const bf16_t* hm = hT + (size_t)m * NC;
    v8f acc[4];
#pragma unroll
    for (int j = 0; j < 4; ++j) acc[j] = (v8f){};
    for (int ks = 0; ks < 32; ++ks) {
        v16bf a = load_frag_row(adj_bf + (size_t)i0 * NDIM + ks * 32, NDIM);
#pragma unroll
        for (int j = 0; j < 4; ++j) {
            v16bf b = load_frag_row(hm + (size_t)(j * 16) * NDIM + ks * 32, NDIM);
            acc[j] = wmma_bf16(a, b, acc[j]);
        }
    }
#pragma unroll
    for (int j = 0; j < 4; ++j)
        store_acc_bf16(hg + (size_t)m * NC + (size_t)i0 * CDIM + j * 16, CDIM, acc[j]);
}

// 11) 1x1 conv: h2[m,n,o] = sum_c hg[m,n,c]*sw[o,c]   -- WMMA, f32 output
//     (sconv_b omitted: a per-channel constant bias cancels exactly in BN mean-sub)
__global__ void k_conv1x1(const bf16_t* __restrict__ hg, const bf16_t* __restrict__ sw_bf,
                          float* __restrict__ h2) {
    const int wave = threadIdx.x >> 5;
    const size_t row0 = ((size_t)blockIdx.x * 8 + wave) * 16;
    v8f acc[4];
#pragma unroll
    for (int j = 0; j < 4; ++j) acc[j] = (v8f){};
#pragma unroll
    for (int ks = 0; ks < 2; ++ks) {
        v16bf a = load_frag_row(hg + row0 * CDIM + ks * 32, CDIM);
#pragma unroll
        for (int j = 0; j < 4; ++j) {
            v16bf b = load_frag_row(sw_bf + j * 16 * 64 + ks * 32, 64);  // B col o = sw row o
            acc[j] = wmma_bf16(a, b, acc[j]);
        }
    }
    const int lane = threadIdx.x & 31;
    const int col = lane & 15, rbase = (lane >> 4) * 8;
#pragma unroll
    for (int j = 0; j < 4; ++j)
#pragma unroll
        for (int rr = 0; rr < 8; ++rr)
            h2[(row0 + rbase + rr) * CDIM + j * 16 + col] = acc[j][rr];
}

// 12) BN partial sums per channel (fixed-order, no fp atomics)
__global__ void k_bn_part(const float* __restrict__ h2, float* __restrict__ part) {
    const int tid = threadIdx.x;
    const size_t base = (size_t)blockIdx.x * 8192;
    float a1 = 0.f, a2 = 0.f;
    for (int e = tid; e < 8192; e += 256) {
        const float v = h2[base + e];
        a1 += v; a2 += v * v;
    }
    __shared__ float r1[256], r2[256];
    r1[tid] = a1; r2[tid] = a2; __syncthreads();
    if (tid < 128) { r1[tid] += r1[tid + 128]; r2[tid] += r2[tid + 128]; }
    __syncthreads();
    if (tid < 64) {   // tid == channel (8192 and 256 are multiples of 64)
        part[((size_t)blockIdx.x * 64 + tid) * 2 + 0] = r1[tid] + r1[tid + 64];
        part[((size_t)blockIdx.x * 64 + tid) * 2 + 1] = r2[tid] + r2[tid + 64];
    }
}

// 13) BN finalize: mean + inv-std per channel
__global__ void k_bn_stats(const float* __restrict__ part, float* __restrict__ stats) {
    const int c = threadIdx.x;                 // 64 threads
    float s1 = 0.f, s2 = 0.f;
    for (int p = 0; p < 768; ++p) {
        s1 += part[((size_t)p * 64 + c) * 2 + 0];
        s2 += part[((size_t)p * 64 + c) * 2 + 1];
    }
    const float cnt = 1.f / (float)(BDIM * NDIM * TDIM);
    const float mean = s1 * cnt;
    const float var  = s2 * cnt - mean * mean;
    stats[c * 2 + 0] = mean;
    stats[c * 2 + 1] = rsqrtf(var + 1e-5f);
}

// 14) BN apply + residual + relu -> out [B,C,N,T] f32
__global__ void k_bn_apply(const float* __restrict__ h2, const float* __restrict__ x2,
                           const float* __restrict__ stats, const float* __restrict__ gamma,
                           const float* __restrict__ beta, float* __restrict__ out) {
    const size_t idx = (size_t)blockIdx.x * 256 + threadIdx.x;  // over B*C*N*T
    const int t = (int)(idx % TDIM);
    size_t q = idx / TDIM;
    const int n = (int)(q % NDIM); q /= NDIM;
    const int c = (int)(q % CDIM);
    const int b = (int)(q / CDIM);
    float v = h2[(((size_t)b * TDIM + t) * NDIM + n) * CDIM + c];
    v = (v - stats[c * 2]) * stats[c * 2 + 1] * gamma[c] + beta[c] + x2[idx];
    out[idx] = fmaxf(v, 0.f);
}

// ---------------------------------------------------------------------------
extern "C" void kernel_launch(void* const* d_in, const int* in_sizes, int n_in,
                              void* d_out, int out_size, void* d_ws, size_t ws_size,
                              hipStream_t stream) {
    (void)in_sizes; (void)n_in; (void)out_size; (void)ws_size;
    const float* x     = (const float*)d_in[0];
    const float* adj   = (const float*)d_in[1];
    const float* W1    = (const float*)d_in[2];
    const float* W2    = (const float*)d_in[3];
    const float* bs    = (const float*)d_in[4];
    const float* Vs    = (const float*)d_in[5];
    const float* U1    = (const float*)d_in[6];
    const float* U3    = (const float*)d_in[7];
    const float* be    = (const float*)d_in[8];
    const float* Ve    = (const float*)d_in[9];
    const float* tcw   = (const float*)d_in[10];
    const float* tcb   = (const float*)d_in[11];
    const float* scw   = (const float*)d_in[12];
    /* sconv_b (d_in[13]) cancels in BN mean subtraction */
    const float* gamma = (const float*)d_in[14];
    const float* beta  = (const float*)d_in[15];
    float* out = (float*)d_out;

    unsigned char* ws = (unsigned char*)d_ws;
    size_t off = 0;
    auto carve = [&](size_t bytes) {
        void* p = ws + off;
        off = (off + bytes + 255) & ~(size_t)255;
        return p;
    };
    float*  lhs    = (float*)carve(MDIM * 4);
    float*  E      = (float*)carve(MDIM * TDIM * 4);
    bf16_t* xatt   = (bf16_t*)carve((size_t)MDIM * NC * 2);   // 12.6 MB
    bf16_t* xattT  = (bf16_t*)carve((size_t)MDIM * NC * 2);   // 12.6 MB
    bf16_t* W1t    = (bf16_t*)carve(64 * 64 * 2);
    bf16_t* W2t    = (bf16_t*)carve(64 * 64 * 2);
    bf16_t* swb    = (bf16_t*)carve(64 * 64 * 2);
    bf16_t* Vs_bf  = (bf16_t*)carve((size_t)NN * 2);          // 2 MB
    bf16_t* adj_bf = (bf16_t*)carve((size_t)NN * 2);          // 2 MB
    bf16_t* lbuf   = (bf16_t*)carve((size_t)MDIM * NC * 2);
    bf16_t* rbuf   = (bf16_t*)carve((size_t)MDIM * NC * 2);
    bf16_t* sigT   = (bf16_t*)carve((size_t)MDIM * NN * 2);   // 192 MB
    bf16_t* S      = (bf16_t*)carve((size_t)MDIM * NN * 2);   // 192 MB
    float*  x2     = (float*)carve((size_t)MDIM * NC * 4);    // 25 MB (residual)
    bf16_t* hT     = (bf16_t*)carve((size_t)MDIM * NC * 2);
    bf16_t* hg     = (bf16_t*)carve((size_t)MDIM * NC * 2);
    float*  h2     = (float*)carve((size_t)MDIM * NC * 4);    // 25 MB
    float*  part   = (float*)carve((size_t)768 * 64 * 2 * 4);
    float*  stats  = (float*)carve(64 * 2 * 4);

    // weight conversions
    hipLaunchKernelGGL(k_f32_to_bf16_T64, dim3(16), dim3(256), 0, stream, W1, W1t);
    hipLaunchKernelGGL(k_f32_to_bf16_T64, dim3(16), dim3(256), 0, stream, W2, W2t);
    hipLaunchKernelGGL(k_f32_to_bf16, dim3(16), dim3(256), 0, stream, scw, swb, 4096);
    hipLaunchKernelGGL(k_f32_to_bf16, dim3(4096), dim3(256), 0, stream, Vs, Vs_bf, NN);
    hipLaunchKernelGGL(k_f32_to_bf16, dim3(4096), dim3(256), 0, stream, adj, adj_bf, NN);

    // temporal attention
    hipLaunchKernelGGL(k_temporal_lhs, dim3(MDIM), dim3(256), 0, stream, x, U3, U1, lhs);
    hipLaunchKernelGGL(k_temporal_E, dim3(BDIM), dim3(160), 0, stream, lhs, be, Ve, E);
    hipLaunchKernelGGL(k_temporal_apply, dim3(2048), dim3(256), 0, stream, x, E, xatt, xattT);

    // spatial attention (WMMA heavy path)
    hipLaunchKernelGGL(k_gemm_lr, dim3(768), dim3(256), 0, stream, xatt, W1t, W2t, lbuf, rbuf);
    hipLaunchKernelGGL(k_prod_sigT, dim3(MDIM, 8, 16), dim3(256), 0, stream, lbuf, rbuf, bs, sigT);
    hipLaunchKernelGGL(k_spatial_S, dim3(MDIM, 8, 16), dim3(256), 0, stream, Vs_bf, sigT, S);
    hipLaunchKernelGGL(k_softmax_S, dim3(MDIM * NDIM), dim3(128), 0, stream, S);
    hipLaunchKernelGGL(k_spatial_apply, dim3(MDIM, 8), dim3(256), 0, stream, S, xattT, x2);

    // temporal conv + graph conv + 1x1 conv
    hipLaunchKernelGGL(k_tconv, dim3(BDIM * NDIM), dim3(256), 0, stream, x2, tcw, tcb, hT);
    hipLaunchKernelGGL(k_graph, dim3(MDIM, 8), dim3(256), 0, stream, adj_bf, hT, hg);
    hipLaunchKernelGGL(k_conv1x1, dim3(768), dim3(256), 0, stream, hg, swb, h2);

    // batchnorm + residual + relu
    hipLaunchKernelGGL(k_bn_part, dim3(768), dim3(256), 0, stream, h2, part);
    hipLaunchKernelGGL(k_bn_stats, dim3(1), dim3(64), 0, stream, part, stats);
    hipLaunchKernelGGL(k_bn_apply, dim3(24576), dim3(256), 0, stream, h2, x2, stats, gamma,
                       beta, out);
}
